// BahdanauAttention_30588757082530
// MI455X (gfx1250) — matrix operand order repaired and checked by reading, not compiled, not run on hardware
//
#include <hip/hip_runtime.h>

#define HD 1024   // hidden
#define SD 2048   // sequence
#define BD 32     // batch
#define STILE 32  // s-rows per workgroup in the WMMA kernel
#define ROWPAD 8                    // bf16 elements of padding per LDS row
#define LDSROW (HD + ROWPAD)        // 1032 halfs = 2064 bytes per row

typedef __attribute__((ext_vector_type(4)))  float        f4;
typedef __attribute__((ext_vector_type(2)))  unsigned int u2;
typedef __attribute__((ext_vector_type(4)))  unsigned int u4;
typedef __attribute__((ext_vector_type(8)))  float        v8f;
typedef __attribute__((ext_vector_type(16))) __bf16       v16bf;

union Frag16 { v16bf v; u4 q[2]; };

__device__ __forceinline__ unsigned int bf16_1(float f) {
    unsigned int u = __float_as_uint(f);
    return (u + 0x7FFFu + ((u >> 16) & 1u)) >> 16;   // round-to-nearest-even
}
__device__ __forceinline__ unsigned int bf16_pack(float lo, float hi) {
    return bf16_1(lo) | (bf16_1(hi) << 16);
}

// ---------------------------------------------------------------- Ua fp32->bf16
__global__ void k_conv_bf16(const float* __restrict__ src,
                            unsigned short* __restrict__ dst) {
    int gid = blockIdx.x * 256 + threadIdx.x;          // HD*HD/4 chunks
    f4 v = ((const f4*)src)[gid];
    u2 p; p.x = bf16_pack(v.x, v.y); p.y = bf16_pack(v.z, v.w);
    *(u2*)(dst + (size_t)gid * 4) = p;
}

// ---------------------------------------------------- pre[b,k] = q.Wa[k]+Wab+Uab
__global__ void k_pre(const float* __restrict__ query, const float* __restrict__ Wa,
                      const float* __restrict__ Wab,   const float* __restrict__ Uab,
                      float* __restrict__ pre) {
    int gid = blockIdx.x * 256 + threadIdx.x;          // b*HD + k
    int b = gid >> 10, k = gid & 1023;
    const f4* q4 = (const f4*)(query + (size_t)b * HD);
    const f4* w4 = (const f4*)(Wa + (size_t)k * HD);
    float acc = 0.f;
#pragma unroll 8
    for (int i = 0; i < HD / 4; i++) {
        f4 a = q4[i], w = w4[i];
        acc += a.x * w.x + a.y * w.y + a.z * w.z + a.w * w.w;
    }
    pre[gid] = acc + Wab[k] + Uab[k];
}

// ------------------------------------- main fused GEMM + tanh + Va dot (WMMA)
__global__ void __launch_bounds__(256)
k_attn(const float* __restrict__ keys, const unsigned short* __restrict__ uabf,
       const float* __restrict__ pre,  const float* __restrict__ vaw,
       const float* __restrict__ vab,  float* __restrict__ scores) {
    extern __shared__ char smem[];
    float* s_part = (float*)smem;                          // 8 waves * 32 rows
    unsigned short* s_keys = (unsigned short*)(smem + 1024);

    const int tid  = threadIdx.x;
    const int lane = tid & 31, wave = tid >> 5;
    const int b  = blockIdx.y;
    const int s0 = blockIdx.x * STILE;
    const int nlo  = lane & 15;     // column in tile / row in 16-group
    const int hsel = lane >> 4;     // which K-half this lane owns

    // stage keys[b, s0..s0+31, :] -> LDS bf16 (padded rows)
    const f4* k4 = (const f4*)keys + (size_t)(b * SD + s0) * (HD / 4);
#pragma unroll 4
    for (int j = 0; j < STILE * (HD / 4) / 256; j++) {
        int idx = j * 256 + tid;
        int row = idx >> 8;         // HD/4 = 256 f4-chunks per row
        int c4  = idx & 255;
        f4 v = k4[row * (HD / 4) + c4];
        u2 p; p.x = bf16_pack(v.x, v.y); p.y = bf16_pack(v.z, v.w);
        *(u2*)(s_keys + row * LDSROW + c4 * 4) = p;
    }
    __syncthreads();

    // per-lane pre-bias and Va values for this wave's 8 n-tiles
    float pre_f[8], va_f[8];
#pragma unroll
    for (int nt = 0; nt < 8; nt++) {
        int k = wave * 128 + nt * 16 + nlo;
        pre_f[nt] = pre[b * HD + k];
        va_f[nt]  = vaw[k];
    }

    v8f acc[2][8];
#pragma unroll
    for (int mt = 0; mt < 2; mt++)
#pragma unroll
        for (int nt = 0; nt < 8; nt++) acc[mt][nt] = (v8f){};

#pragma unroll 1
    for (int ks = 0; ks < HD / 32; ks++) {
        // A fragments (16-bit A 16x32 ISA layout): q[0]=K+0..7 / q[1]=K+16..23
        Frag16 A[2];
#pragma unroll
        for (int mt = 0; mt < 2; mt++) {
            const unsigned short* ap =
                s_keys + (mt * 16 + nlo) * LDSROW + ks * 32 + hsel * 8;
            A[mt].q[0] = *(const u4*)ap;
            A[mt].q[1] = *(const u4*)(ap + 16);
        }
#pragma unroll
        for (int nt = 0; nt < 8; nt++) {
            // B fragment: column = nlo, q[0]=K+0..7 / q[1]=K+16..23 of Ua_w row (matches A)
            const unsigned short* bp =
                uabf + (size_t)(wave * 128 + nt * 16 + nlo) * HD + ks * 32 + hsel * 8;
            Frag16 Bf;
            Bf.q[0] = *(const u4*)bp;
            Bf.q[1] = *(const u4*)(bp + 16);
#pragma unroll
            for (int mt = 0; mt < 2; mt++)
                acc[mt][nt] = __builtin_amdgcn_wmma_f32_16x16x32_bf16(
                    false, A[mt].v, false, Bf.v, (short)0, acc[mt][nt], false, false);
        }
    }

    // epilogue: e = tanh(kproj + pre); partial score += Va*e
    float ps[2][8];
#pragma unroll
    for (int mt = 0; mt < 2; mt++)
#pragma unroll
        for (int r = 0; r < 8; r++) ps[mt][r] = 0.f;

#pragma unroll
    for (int mt = 0; mt < 2; mt++)
#pragma unroll
        for (int nt = 0; nt < 8; nt++) {
            float pr = pre_f[nt], va = va_f[nt];
#pragma unroll
            for (int r = 0; r < 8; r++) {
                float x = acc[mt][nt][r] + pr;
                x = fminf(10.f, fmaxf(-10.f, x));
                float t = __expf(2.f * x);                     // tanh via exp (TRANS)
                float e = (t - 1.f) * __builtin_amdgcn_rcpf(t + 1.f);
                ps[mt][r] += va * e;
            }
        }

    // deterministic cross-lane reduce within each 16-lane group
#pragma unroll
    for (int mt = 0; mt < 2; mt++)
#pragma unroll
        for (int r = 0; r < 8; r++) {
            float v = ps[mt][r];
            v += __shfl_xor(v, 1);
            v += __shfl_xor(v, 2);
            v += __shfl_xor(v, 4);
            v += __shfl_xor(v, 8);
            if (nlo == 0) s_part[wave * 32 + mt * 16 + hsel * 8 + r] = v;
        }
    __syncthreads();

    if (tid < STILE) {
        float sc = vab[0];
#pragma unroll
        for (int w = 0; w < 8; w++) sc += s_part[w * 32 + tid];
        scores[b * SD + s0 + tid] = sc;
    }
}

// ----------------------------------------------------------- per-batch softmax
__global__ void k_softmax(const float* __restrict__ scores, float* __restrict__ out) {
    __shared__ float red[256];
    int b = blockIdx.x, tid = threadIdx.x;
    float v[8];
#pragma unroll
    for (int i = 0; i < 8; i++) v[i] = scores[b * SD + tid + i * 256];
    float m = v[0];
#pragma unroll
    for (int i = 1; i < 8; i++) m = fmaxf(m, v[i]);
    red[tid] = m; __syncthreads();
    for (int off = 128; off > 0; off >>= 1) {
        if (tid < off) red[tid] = fmaxf(red[tid], red[tid + off]);
        __syncthreads();
    }
    m = red[0]; __syncthreads();
    float s = 0.f;
#pragma unroll
    for (int i = 0; i < 8; i++) { v[i] = __expf(v[i] - m); s += v[i]; }
    red[tid] = s; __syncthreads();
    for (int off = 128; off > 0; off >>= 1) {
        if (tid < off) red[tid] += red[tid + off];
        __syncthreads();
    }
    float inv = 1.0f / red[0];
#pragma unroll
    for (int i = 0; i < 8; i++)
        out[BD * HD + b * SD + tid + i * 256] = v[i] * inv;   // weights region
}

// -------------------------------- context = weights . keys  (2-stage, no atomics)
__global__ void k_ctx_part(const float* __restrict__ keys,
                           const float* __restrict__ wts,   // d_out weights region
                           float* __restrict__ part) {
    int b = blockIdx.z, sc = blockIdx.y;
    int h = blockIdx.x * 256 + threadIdx.x;
    const float* kp = keys + ((size_t)b * SD + sc * 256) * HD + h;
    const float* wp = wts + b * SD + sc * 256;
    float acc = 0.f;
#pragma unroll 8
    for (int s = 0; s < 256; s++) acc += wp[s] * kp[(size_t)s * HD];
    part[((size_t)b * 8 + sc) * HD + h] = acc;
}

__global__ void k_ctx_sum(const float* __restrict__ part, float* __restrict__ out) {
    int b = blockIdx.y;
    int h = blockIdx.x * 256 + threadIdx.x;
    float a = 0.f;
#pragma unroll
    for (int sc = 0; sc < 8; sc++) a += part[((size_t)b * 8 + sc) * HD + h];
    out[b * HD + h] = a;   // context region
}

// ------------------------------------------------------------------- launcher
extern "C" void kernel_launch(void* const* d_in, const int* in_sizes, int n_in,
                              void* d_out, int out_size, void* d_ws, size_t ws_size,
                              hipStream_t stream) {
    const float* query = (const float*)d_in[0];
    const float* keys  = (const float*)d_in[1];
    const float* Wa_w  = (const float*)d_in[2];
    const float* Wa_b  = (const float*)d_in[3];
    const float* Ua_w  = (const float*)d_in[4];
    const float* Ua_b  = (const float*)d_in[5];
    const float* Va_w  = (const float*)d_in[6];
    const float* Va_b  = (const float*)d_in[7];
    float* out = (float*)d_out;

    // workspace layout
    char* ws = (char*)d_ws;
    float*          pre    = (float*)ws;                           // 128 KB
    unsigned short* uabf   = (unsigned short*)(ws + 131072);       // 2 MB
    float*          scores = (float*)(ws + 131072 + 2097152);      // 256 KB
    float*          part   = (float*)(ws + 131072 + 2097152 + 262144); // 1 MB

    k_conv_bf16<<<HD * HD / 4 / 256, 256, 0, stream>>>(Ua_w, uabf);
    k_pre<<<BD * HD / 256, 256, 0, stream>>>(query, Wa_w, Wa_b, Ua_b, pre);

    int lds = 1024 + STILE * LDSROW * 2;                           // 67072 B
    static bool attr_done = [&]() {
        (void)hipFuncSetAttribute(reinterpret_cast<const void*>(k_attn),
                                  hipFuncAttributeMaxDynamicSharedMemorySize, lds);
        return true;
    }();
    (void)attr_done;
    (void)hipFuncSetAttribute(reinterpret_cast<const void*>(k_attn),
                              hipFuncAttributeMaxDynamicSharedMemorySize, lds);

    k_attn<<<dim3(SD / STILE, BD), 256, lds, stream>>>(keys, uabf, pre, Va_w, Va_b,
                                                       scores);
    k_softmax<<<BD, 256, 0, stream>>>(scores, out);

    const float* wts = out + BD * HD;   // weights live in d_out
    k_ctx_part<<<dim3(HD / 256, 8, BD), 256, 0, stream>>>(keys, wts, part);
    k_ctx_sum<<<dim3(HD / 256, BD), 256, 0, stream>>>(part, out);
}
